// Model_2130303778821
// MI455X (gfx1250) — matrix operand order repaired and checked by reading, not compiled, not run on hardware
//
#include <hip/hip_runtime.h>
#include <hip/hip_bf16.h>
#include <math.h>
#include <stdint.h>

typedef __bf16 bf16_t;
typedef __attribute__((ext_vector_type(16))) __bf16 v16bf;
typedef __attribute__((ext_vector_type(8)))  float  v8f;
typedef __attribute__((ext_vector_type(8)))  unsigned int v8u;

#define N_ORDER 256
#define PRED_LEN 192
#define SEQ_LEN 768
#define ENC_IN 21
#define BATCH 16
#define MODES 32
#define MROWS 336      // BATCH*ENC_IN
#define MPAD 352       // 22 tiles of 16
#define MT 22
#define NCOLS 16384    // N_ORDER*MODES*2 (re/im interleaved)
#define NT1 1024       // NCOLS/16

// ---------------- WMMA helper ----------------
__device__ __forceinline__ v8f wmma_bf16(v16bf a, v16bf b, v8f c) {
  return __builtin_amdgcn_wmma_f32_16x16x32_bf16(false, a, false, b, (short)0, c, false, false);
}

__device__ __forceinline__ v16bf neg_bf16x16(v16bf x) {
  v8u u;
  __builtin_memcpy(&u, &x, sizeof(u));
  u = u ^ 0x80008000u;            // flip sign bits of packed bf16 pairs
  v16bf r;
  __builtin_memcpy(&r, &u, sizeof(r));
  return r;
}

// ---------------- Stage 0: K_d = B * (A^T)^d, d = 0..L-1 ----------------
__global__ void k_powers(const float* __restrict__ A, const float* __restrict__ Bv,
                         float* __restrict__ Kbuf, int L) {
  __shared__ float sK[N_ORDER];
  int n = threadIdx.x;
  float v = Bv[n];
  sK[n] = v;
  Kbuf[n] = v;
  __syncthreads();
  for (int d = 1; d < L; ++d) {
    const float* Arow = A + (size_t)n * N_ORDER;
    float acc = 0.f;
    #pragma unroll 4
    for (int i = 0; i < N_ORDER; ++i) acc = fmaf(sK[i], Arow[i], acc);
    __syncthreads();
    sK[n] = acc;
    Kbuf[(size_t)d * N_ORDER + n] = acc;
    __syncthreads();
  }
}

// ---------------- Stage 0b: build G in B-fragment-packed bf16 layout ------
// G[s, n] with n = i*64 + k*2 + reim ;  packed as [kc][nt][lane][e]
__device__ __forceinline__ void g_store(bf16_t* __restrict__ G, int s, int n, float v) {
  int kc = s >> 5, w = s & 31;
  int lane = (n & 15) + ((w & 8) << 1);
  int e = (w & 7) + ((w & 16) >> 1);
  size_t idx = (((size_t)kc * NT1 + (n >> 4)) * 32 + lane) * 16 + e;
  G[idx] = (bf16_t)v;
}

__global__ void k_build_G(const float* __restrict__ Kbuf, bf16_t* __restrict__ Gpk, int L) {
  int tid = threadIdx.x;
  int k = tid & 31;                      // mode
  int i = blockIdx.x * 8 + (tid >> 5);   // order index, uniform within a wave
  float c0 = -6.283185307179586f * (float)k / (float)L;
  float Sre = 0.f, Sim = 0.f;
  int n_re = i * 64 + k * 2;
  int n_im = n_re + 1;
  for (int d = 0; d < L; ++d) {
    float Kd = Kbuf[(size_t)d * N_ORDER + i];
    float sd, cd;
    __sincosf(c0 * (float)d, &sd, &cd);
    Sre = fmaf(Kd, cd, Sre);
    Sim = fmaf(Kd, sd, Sim);
    int s = L - 1 - d;
    float sp, cp;
    __sincosf(c0 * (float)s, &sp, &cp);
    g_store(Gpk, s, n_re, Sre * cp - Sim * sp);
    g_store(Gpk, s, n_im, Sre * sp + Sim * cp);
  }
}

// ---------------- pack input window f -> bf16 (MPAD x L, zero-padded) -----
__global__ void k_pack_f(const float* __restrict__ x_enc, bf16_t* __restrict__ F, int L) {
  int idx = blockIdx.x * 256 + threadIdx.x;
  if (idx >= MPAD * L) return;
  int row = idx / L, s = idx - row * L;
  float v = 0.f;
  if (row < MROWS) {
    int b = row / ENC_IN, j = row - b * ENC_IN;
    v = x_enc[((size_t)b * SEQ_LEN + (SEQ_LEN - L + s)) * ENC_IN + j];
  }
  F[idx] = (bf16_t)v;
}

// ---------------- pack Wr/Wi -> bf16 B-fragment layout [k][ic][ot][lane][e]
__global__ void k_packW(const float* __restrict__ Wr, const float* __restrict__ Wi,
                        bf16_t* __restrict__ Pr, bf16_t* __restrict__ Pi) {
  int idx = blockIdx.x * 256 + threadIdx.x;   // 2^21 elements exactly
  int e    = idx & 15;
  int lane = (idx >> 4) & 31;
  int ot   = (idx >> 9) & 15;
  int ic   = (idx >> 13) & 7;
  int k    = (idx >> 16) & 31;
  int i = ic * 32 + ((lane & 16) >> 1) + e + (e & 8);          // B-frag K row
  int o = ot * 16 + (lane & 15);              // B-frag N col
  size_t src = ((size_t)(i * N_ORDER + o)) * MODES + k;
  Pr[idx] = (bf16_t)Wr[src];
  Pi[idx] = (bf16_t)Wi[src];
}

// ---------------- GEMM1: a = f x G  (bf16 WMMA, f32 accum) ----------------
// grid (MT, NT1/8), 256 threads = 8 waves; wave w -> N-tile blockIdx.y*8+w
// Output written in A-fragment-packed layout for GEMM2:
//   Abuf[(((k*2+reim)*8 + ic)*MT + mt)*32*16 + lane2*16 + e2]
__global__ void k_gemm1(const bf16_t* __restrict__ F, const bf16_t* __restrict__ Gpk,
                        bf16_t* __restrict__ Abuf, int L) {
  __shared__ bf16_t sF[16 * SEQ_LEN];
  int mt = blockIdx.x;
  int tid = threadIdx.x;

  // CDNA5 async copy: memory -> LDS without VGPR round-trip (ASYNCcnt path)
  int nbytes = 16 * L * (int)sizeof(bf16_t);
  const char* gsrc = reinterpret_cast<const char*>(F) + (size_t)mt * 16 * L * sizeof(bf16_t);
  for (int ofs = tid * 16; ofs < nbytes; ofs += 256 * 16) {
    unsigned lds_addr = (unsigned)(uintptr_t)(&sF[0]) + (unsigned)ofs;
    unsigned long long gaddr = (unsigned long long)(uintptr_t)(gsrc + ofs);
    asm volatile("global_load_async_to_lds_b128 %0, %1, off"
                 :: "v"(lds_addr), "v"(gaddr) : "memory");
  }
  asm volatile("s_wait_asynccnt 0x0" ::: "memory");
  __syncthreads();

  int wv = tid >> 5, lane = tid & 31;
  int nt = blockIdx.y * 8 + wv;
  int m = lane & 15;
  int akb = (lane & 16) >> 1;   // 0 or 8
  v8f acc = {0.f, 0.f, 0.f, 0.f, 0.f, 0.f, 0.f, 0.f};
  int nchunks = L >> 5;
  for (int kc = 0; kc < nchunks; ++kc) {
    v16bf av, bv;
    #pragma unroll
    for (int e = 0; e < 16; ++e) {
      int kk = kc * 32 + akb + e + (e & 8);   // A-frag K index per ISA table
      av[e] = sF[m * L + kk];
    }
    const bf16_t* gp = Gpk + ((((size_t)kc * NT1 + nt) * 32 + lane) << 4);
    bv = *reinterpret_cast<const v16bf*>(gp);
    if (kc + 1 < nchunks)
      __builtin_prefetch(Gpk + ((((size_t)(kc + 1) * NT1 + nt) * 32 + lane) << 4), 0, 1);
    acc = wmma_bf16(av, bv, acc);
  }
  // D layout: lane -> N=lane&15, M = (lane<16?0:8)+r ; scatter into A-frag layout
  int ncol = nt * 16 + (lane & 15);
  int i = ncol >> 6, rem = ncol & 63;
  int kmode = rem >> 1, reim = rem & 1;
  int ic = i >> 5, i_in = i & 31;
  int lane_hi = (i_in >> 3) & 1;                       // 1 -> lanes 16..31
  int e2 = (i_in & 7) + ((i_in >> 4) & 1) * 8;        // element within fragment
  int mrow0 = mt * 16 + ((lane & 16) >> 1);
  size_t base = ((((size_t)(kmode * 2 + reim) * 8 + ic) * MT + mt) * 32) * 16;
  #pragma unroll
  for (int r = 0; r < 8; ++r) {
    int mm = mrow0 + r;
    int lane2 = (mm & 15) + (lane_hi << 4);
    Abuf[base + (size_t)lane2 * 16 + e2] = (bf16_t)acc[r];
  }
}

// ---------------- GEMM2: per-mode complex mix + fused irfft@t=191 ---------
// grid (MT, 4), 128 threads = 4 waves; wave w -> O-tile blockIdx.y*4+w
__global__ void k_gemm2(const bf16_t* __restrict__ Abuf, const bf16_t* __restrict__ Pr,
                        const bf16_t* __restrict__ Pi, float* __restrict__ Xdec, int L) {
  int tid = threadIdx.x, wv = tid >> 5, lane = tid & 31;
  int mt = blockIdx.x, ot = blockIdx.y * 4 + wv;
  float fin[8];
  #pragma unroll
  for (int q = 0; q < 8; ++q) fin[q] = 0.f;
  float invL = 1.0f / (float)L;

  for (int k = 0; k < MODES; ++k) {
    v8f re = {0.f, 0.f, 0.f, 0.f, 0.f, 0.f, 0.f, 0.f};
    v8f im = {0.f, 0.f, 0.f, 0.f, 0.f, 0.f, 0.f, 0.f};
    for (int ic = 0; ic < 8; ++ic) {
      // A-fragments: single 32B vector load each (packed by GEMM1)
      size_t abase_r = ((((size_t)(k * 2 + 0) * 8 + ic) * MT + mt) * 32 + lane) << 4;
      size_t abase_i = ((((size_t)(k * 2 + 1) * 8 + ic) * MT + mt) * 32 + lane) << 4;
      v16bf ar = *reinterpret_cast<const v16bf*>(Abuf + abase_r);
      v16bf ai = *reinterpret_cast<const v16bf*>(Abuf + abase_i);
      size_t widx = ((((size_t)k * 8 + ic) * 16 + ot) * 32 + lane) << 4;
      v16bf wr = *reinterpret_cast<const v16bf*>(Pr + widx);
      v16bf wi = *reinterpret_cast<const v16bf*>(Pi + widx);
      v16bf wni = neg_bf16x16(wi);          // f16/bf16 WMMA has no A/B NEG (ISA 7.12)
      re = wmma_bf16(ar, wr, re);
      re = wmma_bf16(ai, wni, re);          // Re += -aI*Wi
      im = wmma_bf16(ar, wi, im);
      im = wmma_bf16(ai, wr, im);
    }
    // irfft at t = PRED_LEN-1: x = (1/L)[X0R + sum 2(Re cos - Im sin)]
    float ph = 6.283185307179586f * (float)k * (float)(PRED_LEN - 1) / (float)L;
    float sp, cp;
    __sincosf(ph, &sp, &cp);
    float scale = (k == 0 ? 1.0f : 2.0f) * invL;
    float cw = scale * cp, sw = scale * sp;
    #pragma unroll
    for (int q = 0; q < 8; ++q) fin[q] += cw * re[q] - sw * im[q];
  }
  int mrow0 = mt * 16 + ((lane & 16) >> 1);
  int o = ot * 16 + (lane & 15);
  #pragma unroll
  for (int r = 0; r < 8; ++r)
    Xdec[(size_t)(mrow0 + r) * N_ORDER + o] = fin[r];
}

// ---------------- Final: Legendre projection + 3-scale MLP mix ------------
__global__ void k_final(const float* __restrict__ xd0, const float* __restrict__ xd1,
                        const float* __restrict__ xd2,
                        const float* __restrict__ E0, const float* __restrict__ E1,
                        const float* __restrict__ E2,
                        const float* __restrict__ mlp_w, const float* __restrict__ mlp_b,
                        float* __restrict__ out) {
  int row = blockIdx.x;        // 0..335
  int t = threadIdx.x;         // 0..191
  const float* xds[3] = {xd0, xd1, xd2};
  const float* Es[3]  = {E0, E1, E2};
  const int   Ls[3]   = {PRED_LEN, 2 * PRED_LEN, 4 * PRED_LEN};
  float acc = mlp_b[0];
  for (int s = 0; s < 3; ++s) {
    const float* xr = xds[s] + (size_t)row * N_ORDER;
    const float* er = Es[s] + (size_t)(Ls[s] - PRED_LEN + t) * N_ORDER;
    float a2 = 0.f;
    #pragma unroll 4
    for (int o = 0; o < N_ORDER; ++o) a2 = fmaf(xr[o], er[o], a2);
    acc = fmaf(mlp_w[s], a2, acc);
  }
  int b = row / ENC_IN, j = row - b * ENC_IN;
  out[((size_t)b * PRED_LEN + t) * ENC_IN + j] = acc;
}

// ---------------- Host orchestration ----------------
extern "C" void kernel_launch(void* const* d_in, const int* in_sizes, int n_in,
                              void* d_out, int out_size, void* d_ws, size_t ws_size,
                              hipStream_t stream) {
  (void)in_sizes; (void)n_in; (void)out_size; (void)ws_size;
  const float* x_enc = (const float*)d_in[0];
  const float* mlp_w = (const float*)d_in[16];
  const float* mlp_b = (const float*)d_in[17];

  char* ws = (char*)d_ws;
  // workspace layout (bytes), buffers reused across the 3 sequential scales
  const size_t OFF_K   = 0;                          // 768*256*4      = 786432
  const size_t OFF_GPK = OFF_K   + 786432;           // 768*16384*2    = 25165824
  const size_t OFF_F   = OFF_GPK + 25165824;         // 352*768*2      = 540672
  const size_t OFF_A   = OFF_F   + 540672;           // 352*16384*2    = 11534336
  const size_t OFF_PR  = OFF_A   + 11534336;         // 2^21*2         = 4194304
  const size_t OFF_PI  = OFF_PR  + 4194304;          // 4194304
  const size_t OFF_XD  = OFF_PI  + 4194304;          // 3*352*256*4    = 1081344

  float*  Kbuf = (float*)(ws + OFF_K);
  bf16_t* Gpk  = (bf16_t*)(ws + OFF_GPK);
  bf16_t* Fb   = (bf16_t*)(ws + OFF_F);
  bf16_t* Abuf = (bf16_t*)(ws + OFF_A);
  bf16_t* Pr   = (bf16_t*)(ws + OFF_PR);
  bf16_t* Pi   = (bf16_t*)(ws + OFF_PI);
  float*  Xd   = (float*)(ws + OFF_XD);

  for (int s = 0; s < 3; ++s) {
    int L = PRED_LEN << s;
    const float* A  = (const float*)d_in[1 + 5 * s];
    const float* Bv = (const float*)d_in[2 + 5 * s];
    const float* Wr = (const float*)d_in[4 + 5 * s];
    const float* Wi = (const float*)d_in[5 + 5 * s];
    float* Xdec = Xd + (size_t)s * MPAD * N_ORDER;

    k_powers<<<1, 256, 0, stream>>>(A, Bv, Kbuf, L);
    k_build_G<<<32, 256, 0, stream>>>(Kbuf, Gpk, L);
    k_pack_f<<<(MPAD * L + 255) / 256, 256, 0, stream>>>(x_enc, Fb, L);
    k_packW<<<8192, 256, 0, stream>>>(Wr, Wi, Pr, Pi);
    dim3 g1(MT, NT1 / 8);
    k_gemm1<<<g1, 256, 0, stream>>>(Fb, Gpk, Abuf, L);
    dim3 g2(MT, 4);
    k_gemm2<<<g2, 128, 0, stream>>>(Abuf, Pr, Pi, Xdec, L);
  }

  const float* E0 = (const float*)d_in[3];
  const float* E1 = (const float*)d_in[8];
  const float* E2 = (const float*)d_in[13];
  k_final<<<MROWS, PRED_LEN, 0, stream>>>(Xd, Xd + (size_t)MPAD * N_ORDER,
                                          Xd + (size_t)2 * MPAD * N_ORDER,
                                          E0, E1, E2, mlp_w, mlp_b, (float*)d_out);
}
